// VAE_Decoder_48146583388741
// MI455X (gfx1250) — matrix operand order repaired and checked by reading, not compiled, not run on hardware
//
#include <hip/hip_runtime.h>

typedef __attribute__((ext_vector_type(16))) _Float16 v16h;
typedef __attribute__((ext_vector_type(8)))  float    v8f;

#define S_DIM 256
#define B_DIM 64
#define H_DIM 1024
#define R_DIM 512
#define V_DIM 512

// ---------- WMMA fragment helpers (CDNA5 16x16x32 f16 layouts, ISA 7.12.2) ----------

// A (16x32 f16, row-major, stride lda): lane L -> row M=L&15.
// elems 0..7 = K = (L>>4)*8 + 0..7 ; elems 8..15 = K = 16 + (L>>4)*8 + 0..7
__device__ __forceinline__ v16h load_a_frag(const _Float16* __restrict__ A, int lda) {
  int lane = threadIdx.x & 31;
  const _Float16* p = A + (lane & 15) * lda + ((lane >> 4) << 3);
  union { uint4 u[2]; v16h v; } t;
  t.u[0] = *(const uint4*)p;
  t.u[1] = *(const uint4*)(p + 16);
  return t.v;
}

// B (32x16) built from a row-major weight matrix W (N x K, K contiguous, stride ldw):
// B[k][n] = W[n][k].  lane L -> col N=L&15, elems 0..7 -> K = (L>>4)*8 + e ; elems 8..15 -> K = 16 + (L>>4)*8 + (e-8)
__device__ __forceinline__ v16h load_bt_frag(const _Float16* __restrict__ W, int ldw) {
  int lane = threadIdx.x & 31;
  const _Float16* p = W + (lane & 15) * ldw + ((lane >> 4) << 3);
  union { uint4 u[2]; v16h v; } t;
  t.u[0] = *(const uint4*)p;
  t.u[1] = *(const uint4*)(p + 16);
  return t.v;
}

__device__ __forceinline__ v8f wmma_f16(v16h a, v16h b, v8f c) {
  // (neg_a, A, neg_b, B, c_mod, C, reuse_a, reuse_b)
  return __builtin_amdgcn_wmma_f32_16x16x32_f16(false, a, false, b, (short)0, c, false, false);
}

__device__ __forceinline__ float sigmoidf_(float x) {
  return 1.0f / (1.0f + __expf(-x));
}
__device__ __forceinline__ float tanhf_(float x) {
  x = fminf(fmaxf(x, -15.0f), 15.0f);
  float e = __expf(2.0f * x);
  return (e - 1.0f) / (e + 1.0f);
}

// ---------- utility kernels ----------

__global__ void cvt_f32_f16(const float* __restrict__ src, _Float16* __restrict__ dst, int n4) {
  int i = blockIdx.x * blockDim.x + threadIdx.x;
  if (i < n4) {
    float4 f = ((const float4*)src)[i];
    union { _Float16 h[4]; uint2 u; } t;
    t.h[0] = (_Float16)f.x; t.h[1] = (_Float16)f.y;
    t.h[2] = (_Float16)f.z; t.h[3] = (_Float16)f.w;
    *(uint2*)(dst + 4 * i) = t.u;
  }
}

__global__ void lstm_init(const int* __restrict__ cond, const float* __restrict__ emb,
                          float* __restrict__ h32, float* __restrict__ c32,
                          _Float16* __restrict__ h16) {
  int i = blockIdx.x * blockDim.x + threadIdx.x;   // 0 .. B*H-1
  int b = i >> 10;                                 // H = 1024
  int j = i & (H_DIM - 1);
  float v = emb[cond[b] * H_DIM + j];
  h32[i] = v;
  c32[i] = v;
  h16[i] = (_Float16)v;
}

// ---------- fused LSTM timestep: gates GEMM (WMMA) + one-hot gather + cell ----------
// 256 waves: mt = wave & 3 (16-batch tile), jt = wave >> 2 (16-wide j slice).
// Consecutive waves (same workgroup) share jt's W_hh fragments -> WGP$ hits.
// Each wave computes the 4 gate tiles (i,f,g,o) for its (b,j) block sharing the A
// fragment, software-pipelined (prefetch k+32 while WMMAing k), then finishes the
// LSTM cell elementwise fully in-register.
__global__ __launch_bounds__(256) void lstm_step(
    const _Float16* __restrict__ Whh16,   // (4H, H) f16, K contiguous
    const _Float16* __restrict__ h16_in,  // (B, H) f16
    _Float16* __restrict__ h16_out,       // (B, H) f16
    const int*  __restrict__ e_t,         // (B,)
    const float* __restrict__ W_ih,       // (4H, R) f32
    const float* __restrict__ b_ih,       // (4H,)
    const float* __restrict__ b_hh,       // (4H,)
    float* __restrict__ h32,              // (B, H) live h (d_out region)
    float* __restrict__ c32,              // (B, H) live c (d_out region)
    _Float16* __restrict__ out_t)         // outs16 + t*B*H
{
  int wave = (blockIdx.x * blockDim.x + threadIdx.x) >> 5;
  int mt = wave & 3;
  int jt = wave >> 2;
  int lane = threadIdx.x & 31;

  v8f acc0 = {}, acc1 = {}, acc2 = {}, acc3 = {};
  const _Float16* Abase = h16_in + (size_t)(mt << 4) * H_DIM;
  const _Float16* B0 = Whh16 + (size_t)(0 * H_DIM + (jt << 4)) * H_DIM;
  const _Float16* B1 = Whh16 + (size_t)(1 * H_DIM + (jt << 4)) * H_DIM;
  const _Float16* B2 = Whh16 + (size_t)(2 * H_DIM + (jt << 4)) * H_DIM;
  const _Float16* B3 = Whh16 + (size_t)(3 * H_DIM + (jt << 4)) * H_DIM;

  // software pipeline: stage 0
  v16h a  = load_a_frag(Abase, H_DIM);
  v16h b0 = load_bt_frag(B0, H_DIM);
  v16h b1 = load_bt_frag(B1, H_DIM);
  v16h b2 = load_bt_frag(B2, H_DIM);
  v16h b3 = load_bt_frag(B3, H_DIM);

  for (int k = 32; k < H_DIM; k += 32) {
    v16h an  = load_a_frag(Abase + k, H_DIM);
    v16h bn0 = load_bt_frag(B0 + k, H_DIM);
    v16h bn1 = load_bt_frag(B1 + k, H_DIM);
    v16h bn2 = load_bt_frag(B2 + k, H_DIM);
    v16h bn3 = load_bt_frag(B3 + k, H_DIM);
    acc0 = wmma_f16(a, b0, acc0);
    acc1 = wmma_f16(a, b1, acc1);
    acc2 = wmma_f16(a, b2, acc2);
    acc3 = wmma_f16(a, b3, acc3);
    a = an; b0 = bn0; b1 = bn1; b2 = bn2; b3 = bn3;
  }
  acc0 = wmma_f16(a, b0, acc0);
  acc1 = wmma_f16(a, b1, acc1);
  acc2 = wmma_f16(a, b2, acc2);
  acc3 = wmma_f16(a, b3, acc3);

  int jj = (jt << 4) + (lane & 15);                // j in [0,H)
  float bi = b_ih[0 * H_DIM + jj] + b_hh[0 * H_DIM + jj];
  float bf = b_ih[1 * H_DIM + jj] + b_hh[1 * H_DIM + jj];
  float bg = b_ih[2 * H_DIM + jj] + b_hh[2 * H_DIM + jj];
  float bo = b_ih[3 * H_DIM + jj] + b_hh[3 * H_DIM + jj];
  int brow = (mt << 4) + ((lane >> 4) << 3);

#pragma unroll
  for (int vv = 0; vv < 8; ++vv) {
    int b = brow + vv;
    int e = e_t[b];
    const float* Wc = W_ih + e;                    // one-hot input: gather column e
    float gi = acc0[vv] + bi + Wc[(0 * H_DIM + jj) * R_DIM];
    float gf = acc1[vv] + bf + Wc[(1 * H_DIM + jj) * R_DIM];
    float gg = acc2[vv] + bg + Wc[(2 * H_DIM + jj) * R_DIM];
    float go = acc3[vv] + bo + Wc[(3 * H_DIM + jj) * R_DIM];
    int idx = b * H_DIM + jj;
    float cn = sigmoidf_(gf) * c32[idx] + sigmoidf_(gi) * tanhf_(gg);
    float hn = sigmoidf_(go) * tanhf_(cn);
    c32[idx] = cn;
    h32[idx] = hn;
    _Float16 hh = (_Float16)hn;
    h16_out[idx] = hh;
    out_t[idx] = hh;
  }
}

// ---------- final FC: logits = outs @ fc_w.T + fc_b (16384 x 512, K = 1024) ----------
// 8192 waves; nt = wave >> 8 so a workgroup's 8 waves share the fc_w fragments
// (WGP$ hits); each wave: 1 N-tile x 4 M-tiles, software-pipelined.
__global__ __launch_bounds__(256) void fc_logits(
    const _Float16* __restrict__ outs16,  // (S*B, H) f16
    const _Float16* __restrict__ fcw16,   // (V, H) f16
    const float* __restrict__ fc_b,       // (V,)
    float* __restrict__ logits)           // (S*B, V)
{
  int wave = (blockIdx.x * blockDim.x + threadIdx.x) >> 5;
  int mg = wave & 255;       // 256 groups of 4 M-tiles
  int nt = wave >> 8;        // 32 N-tiles
  int lane = threadIdx.x & 31;

  v8f acc0 = {}, acc1 = {}, acc2 = {}, acc3 = {};
  const _Float16* Bb = fcw16 + (size_t)(nt << 4) * H_DIM;
  const _Float16* A0 = outs16 + (size_t)(mg * 64) * H_DIM;
  const _Float16* A1 = A0 + (size_t)16 * H_DIM;
  const _Float16* A2 = A0 + (size_t)32 * H_DIM;
  const _Float16* A3 = A0 + (size_t)48 * H_DIM;

  v16h bfrag = load_bt_frag(Bb, H_DIM);
  v16h a0 = load_a_frag(A0, H_DIM);
  v16h a1 = load_a_frag(A1, H_DIM);
  v16h a2 = load_a_frag(A2, H_DIM);
  v16h a3 = load_a_frag(A3, H_DIM);

  for (int k = 32; k < H_DIM; k += 32) {
    v16h bn  = load_bt_frag(Bb + k, H_DIM);
    v16h an0 = load_a_frag(A0 + k, H_DIM);
    v16h an1 = load_a_frag(A1 + k, H_DIM);
    v16h an2 = load_a_frag(A2 + k, H_DIM);
    v16h an3 = load_a_frag(A3 + k, H_DIM);
    acc0 = wmma_f16(a0, bfrag, acc0);
    acc1 = wmma_f16(a1, bfrag, acc1);
    acc2 = wmma_f16(a2, bfrag, acc2);
    acc3 = wmma_f16(a3, bfrag, acc3);
    bfrag = bn; a0 = an0; a1 = an1; a2 = an2; a3 = an3;
  }
  acc0 = wmma_f16(a0, bfrag, acc0);
  acc1 = wmma_f16(a1, bfrag, acc1);
  acc2 = wmma_f16(a2, bfrag, acc2);
  acc3 = wmma_f16(a3, bfrag, acc3);

  int col = (nt << 4) + (lane & 15);
  float bias = fc_b[col];
  int r0 = mg * 64 + ((lane >> 4) << 3);
#pragma unroll
  for (int vv = 0; vv < 8; ++vv) {
    logits[(size_t)(r0 + vv) * V_DIM + col]      = acc0[vv] + bias;
    logits[(size_t)(r0 + 16 + vv) * V_DIM + col] = acc1[vv] + bias;
    logits[(size_t)(r0 + 32 + vv) * V_DIM + col] = acc2[vv] + bias;
    logits[(size_t)(r0 + 48 + vv) * V_DIM + col] = acc3[vv] + bias;
  }
}

// ---------- log-softmax NLL: one wave32 per row (512 cols -> 16 per lane) ----------
__global__ __launch_bounds__(256) void loss_rows_k(
    const float* __restrict__ logits, const int* __restrict__ labels,
    float* __restrict__ rloss)
{
  int row = (blockIdx.x * blockDim.x + threadIdx.x) >> 5;
  int lane = threadIdx.x & 31;
  const float* r = logits + (size_t)row * V_DIM;
  float vals[16];
  float mx = -1e30f;
#pragma unroll
  for (int i = 0; i < 16; ++i) {
    vals[i] = r[lane + (i << 5)];
    mx = fmaxf(mx, vals[i]);
  }
  for (int off = 16; off; off >>= 1) mx = fmaxf(mx, __shfl_xor(mx, off, 32));
  float s = 0.0f;
#pragma unroll
  for (int i = 0; i < 16; ++i) s += __expf(vals[i] - mx);
  for (int off = 16; off; off >>= 1) s += __shfl_xor(s, off, 32);
  if (lane == 0) {
    float lv = r[labels[row]];
    rloss[row] = -(lv - mx - __logf(s));
  }
}

__global__ void loss_reduce_k(const float* __restrict__ rloss, float* __restrict__ out) {
  __shared__ float sm[256];
  float s = 0.0f;
  for (int i = threadIdx.x; i < S_DIM * B_DIM; i += 256) s += rloss[i];
  sm[threadIdx.x] = s;
  __syncthreads();
  for (int off = 128; off; off >>= 1) {
    if ((int)threadIdx.x < off) sm[threadIdx.x] += sm[threadIdx.x + off];
    __syncthreads();
  }
  if (threadIdx.x == 0) out[0] = sm[0] / (float)(S_DIM * B_DIM);
}

// ---------- host ----------

extern "C" void kernel_launch(void* const* d_in, const int* in_sizes, int n_in,
                              void* d_out, int out_size, void* d_ws, size_t ws_size,
                              hipStream_t stream) {
  (void)in_sizes; (void)n_in; (void)out_size; (void)ws_size;
  const int*   cond = (const int*)d_in[1];
  const int*   e_in = (const int*)d_in[2];
  const float* emb  = (const float*)d_in[4];
  const float* W_ih = (const float*)d_in[5];
  const float* W_hh = (const float*)d_in[6];
  const float* b_ih = (const float*)d_in[7];
  const float* b_hh = (const float*)d_in[8];
  const float* fc_w = (const float*)d_in[9];
  const float* fc_b = (const float*)d_in[10];

  char* ws = (char*)d_ws;
  size_t off = 0;
  _Float16* Whh16  = (_Float16*)(ws + off); off += (size_t)4 * H_DIM * H_DIM * 2;
  _Float16* fcw16  = (_Float16*)(ws + off); off += (size_t)V_DIM * H_DIM * 2;
  _Float16* h16a   = (_Float16*)(ws + off); off += (size_t)B_DIM * H_DIM * 2;
  _Float16* h16b   = (_Float16*)(ws + off); off += (size_t)B_DIM * H_DIM * 2;
  _Float16* outs16 = (_Float16*)(ws + off); off += (size_t)S_DIM * B_DIM * H_DIM * 2;
  float*    rloss  = (float*)(ws + off);    off += (size_t)S_DIM * B_DIM * 4;

  float* logits = (float*)d_out;
  float* h32    = logits + (size_t)S_DIM * B_DIM * V_DIM;
  float* c32    = h32 + (size_t)B_DIM * H_DIM;
  float* lossp  = c32 + (size_t)B_DIM * H_DIM;

  // One-time per-call f16 conversions of the weight matrices (L2-resident afterwards).
  {
    int n4 = 4 * H_DIM * H_DIM / 4;
    cvt_f32_f16<<<(n4 + 255) / 256, 256, 0, stream>>>(W_hh, Whh16, n4);
  }
  {
    int n4 = V_DIM * H_DIM / 4;
    cvt_f32_f16<<<(n4 + 255) / 256, 256, 0, stream>>>(fc_w, fcw16, n4);
  }

  lstm_init<<<B_DIM * H_DIM / 256, 256, 0, stream>>>(cond, emb, h32, c32, h16a);

  for (int t = 0; t < S_DIM; ++t) {
    const _Float16* hin = (t & 1) ? h16b : h16a;
    _Float16*      hout = (t & 1) ? h16a : h16b;
    lstm_step<<<32, 256, 0, stream>>>(Whh16, hin, hout, e_in + t * B_DIM,
                                      W_ih, b_ih, b_hh, h32, c32,
                                      outs16 + (size_t)t * B_DIM * H_DIM);
  }

  fc_logits<<<1024, 256, 0, stream>>>(outs16, fcw16, fc_b, logits);
  loss_rows_k<<<S_DIM * B_DIM / 8, 256, 0, stream>>>(logits, e_in, rloss);
  loss_reduce_k<<<1, 256, 0, stream>>>(rloss, lossp);
}
